// MAB_61607010894006
// MI455X (gfx1250) — matrix operand order repaired and checked by reading, not compiled, not run on hardware
//
#include <hip/hip_runtime.h>
#include <hip/hip_bf16.h>

typedef __attribute__((ext_vector_type(16))) _Float16 v16h;
typedef __attribute__((ext_vector_type(8)))  float    v8f;
typedef __attribute__((ext_vector_type(8)))  _Float16 v8h;
typedef __attribute__((ext_vector_type(4)))  _Float16 v4h;
typedef __attribute__((ext_vector_type(4)))  int      v4i;

#define B_SZ 8
#define NQ   1024
#define NK   1024
#define DIM  512
#define NH   8
#define DH   64
#define NEGV (-1e9f)

#if __has_builtin(__builtin_amdgcn_global_load_async_to_lds_b128) && \
    __has_builtin(__builtin_amdgcn_s_wait_asynccnt)
#define HAVE_ASYNC 1
#else
#define HAVE_ASYNC 0
#endif

typedef __attribute__((address_space(1))) v4i gv4i; // global
typedef __attribute__((address_space(3))) v4i lv4i; // LDS

__device__ __forceinline__ void async_copy_b128(const _Float16* gsrc, _Float16* ldst) {
#if HAVE_ASYNC
    __builtin_amdgcn_global_load_async_to_lds_b128(
        (gv4i*)gsrc, (lv4i*)ldst, 0, 0);
#else
    *(v8h*)ldst = *(const v8h*)gsrc;
#endif
}

__device__ __forceinline__ void wait_async() {
#if HAVE_ASYNC
    __builtin_amdgcn_s_wait_asynccnt(0);
#endif
}

// ---------------------------------------------------------------------------
// 16-lane-half butterfly on the VALU via v_permlane16_b32.
// ---------------------------------------------------------------------------
__device__ __forceinline__ float permx16(float v, unsigned s0, unsigned s1) {
#if __has_builtin(__builtin_amdgcn_permlane16)
    int i = __float_as_int(v);
    i = __builtin_amdgcn_permlane16(i, i, s0, s1, false, false);
    return __int_as_float(i);
#else
    (void)s0; (void)s1;
    return 0.0f;
#endif
}

__device__ __forceinline__ float half16_max(float v) {
#if __has_builtin(__builtin_amdgcn_permlane16)
    v = fmaxf(v, permx16(v, 0x67452301u, 0xefcdab89u));
    v = fmaxf(v, permx16(v, 0x45670123u, 0xcdef89abu));
    v = fmaxf(v, permx16(v, 0x01234567u, 0x89abcdefu));
    v = fmaxf(v, permx16(v, 0xfedcba98u, 0x76543210u));
#else
#pragma unroll
    for (int off = 1; off < 16; off <<= 1) v = fmaxf(v, __shfl_xor(v, off, 32));
#endif
    return v;
}

__device__ __forceinline__ float half16_sum(float v) {
#if __has_builtin(__builtin_amdgcn_permlane16)
    v += permx16(v, 0x67452301u, 0xefcdab89u);
    v += permx16(v, 0x45670123u, 0xcdef89abu);
    v += permx16(v, 0x01234567u, 0x89abcdefu);
    v += permx16(v, 0xfedcba98u, 0x76543210u);
#else
#pragma unroll
    for (int off = 1; off < 16; off <<= 1) v += __shfl_xor(v, off, 32);
#endif
    return v;
}

// ---------------------------------------------------------------------------
// One-time weight prep: f32 [K][N] -> f16 transposed [N][K]
// ---------------------------------------------------------------------------
__global__ void cvt_transpose_f16(const float* __restrict__ src, _Float16* __restrict__ dst) {
    int idx = blockIdx.x * blockDim.x + threadIdx.x;
    int k = idx >> 9, n = idx & 511;
    dst[(size_t)n * DIM + k] = (_Float16)src[idx];
}

// ---------------------------------------------------------------------------
// GEMM with output modes:
//  MODE 0: C  = A@W + bias                  (f32, q projection)
//  MODE 1: C  = relu(A@W + bias) + X        (f32, output projection)
//  MODE 2: C16= A@W + bias                  (f16 row-major, k projection)
//  MODE 3: C16= A@W + bias, per-head T      (f16 [b][h][dh][nk], v projection)
// Block tile 128x128, K-step 32, 8 waves, wave tile 32x64 (8 WMMAs/step).
// A-tile register-pipelined; B-tile (f16 weights) DMA'd with async-to-LDS
// into a double buffer overlapping the WMMA section.
// ---------------------------------------------------------------------------
template <int MODE>
__global__ __launch_bounds__(256)
void gemm512(const float* __restrict__ A, const _Float16* __restrict__ WT,
             const float* __restrict__ bias, const float* __restrict__ X,
             float* __restrict__ C, _Float16* __restrict__ C16) {
    constexpr int TM = 128, TN = 128, TK = 32;
    __shared__ __attribute__((aligned(16))) _Float16 As [TM][40];     // stride 80B
    __shared__ __attribute__((aligned(16))) _Float16 BsT[2][TN][56];  // stride 112B (16B mult)

    const int tid  = threadIdx.x;
    const int wave = tid >> 5;
    const int lane = tid & 31;
    const int wm = (wave & 3) * 32;
    const int wn = (wave >> 2) * 64;
    const int m0 = blockIdx.x * TM;
    const int n0 = blockIdx.y * TN;

    const int ml  = lane & 15;
    const int akb = (lane < 16) ? 0 : 8;
    const int bkb = (lane < 16) ? 0 : 8;
    const int rb  = (lane < 16) ? 0 : 8;

    // A staging coords: 4 x float4 per thread
    int rA[4], cA[4];
#pragma unroll
    for (int i = 0; i < 4; ++i) {
        int e = (tid + i * 256) * 4;
        rA[i] = e >> 5; cA[i] = e & 31;
    }
    // B staging coords: 2 x b128 per thread (128 rows x 32 k = 512 segs)
    int rB[2], sB[2];
#pragma unroll
    for (int i = 0; i < 2; ++i) {
        int idx = tid + i * 256;
        rB[i] = idx >> 2; sB[i] = (idx & 3) * 8;
    }

    float4 aReg[4];
    auto load_A = [&](int k0) {
#pragma unroll
        for (int i = 0; i < 4; ++i)
            aReg[i] = *(const float4*)(A + (size_t)(m0 + rA[i]) * DIM + k0 + cA[i]);
    };
    auto store_A = [&]() {
#pragma unroll
        for (int i = 0; i < 4; ++i) {
            v4h hv = { (_Float16)aReg[i].x, (_Float16)aReg[i].y,
                       (_Float16)aReg[i].z, (_Float16)aReg[i].w };
            *(v4h*)&As[rA[i]][cA[i]] = hv;
        }
    };
    auto stage_B = [&](int k0, int buf) {
#pragma unroll
        for (int i = 0; i < 2; ++i)
            async_copy_b128(WT + (size_t)(n0 + rB[i]) * DIM + k0 + sB[i],
                            &BsT[buf][rB[i]][sB[i]]);
    };

    v8f acc[2][4] = {};

    stage_B(0, 0);
    load_A(0);
    store_A();
    wait_async();
    __syncthreads();

    int buf = 0;
    for (int k0 = 0; k0 < DIM; k0 += TK) {
        const bool more = (k0 + TK) < DIM;
        if (more) { stage_B(k0 + TK, buf ^ 1); load_A(k0 + TK); }

        v16h af[2], bf[4];
#pragma unroll
        for (int t = 0; t < 2; ++t) {
            const _Float16* ar = &As[wm + t * 16 + ml][0];
#pragma unroll
            for (int j = 0; j < 4; ++j) {
                af[t][2 * j]         = ar[akb + 2 * j];
                af[t][2 * j + 1]     = ar[akb + 2 * j + 1];
                af[t][8 + 2 * j]     = ar[akb + 16 + 2 * j];
                af[t][8 + 2 * j + 1] = ar[akb + 16 + 2 * j + 1];
            }
        }
#pragma unroll
        for (int n = 0; n < 4; ++n) {
            const _Float16* br = &BsT[buf][wn + n * 16 + ml][0];
#pragma unroll
            for (int j = 0; j < 8; ++j) {
                bf[n][2 * j]     = br[bkb + 2 * j + (j >= 4 ? 8 : 0)];
                bf[n][2 * j + 1] = br[bkb + 2 * j + 1 + (j >= 4 ? 8 : 0)];
            }
        }
#pragma unroll
        for (int i = 0; i < 2; ++i)
#pragma unroll
            for (int n = 0; n < 4; ++n)
                acc[i][n] = __builtin_amdgcn_wmma_f32_16x16x32_f16(
                    false, af[i], false, bf[n], (short)0, acc[i][n], false, false);

        __syncthreads();                 // all waves done reading As / BsT[buf]
        if (more) store_A();
        wait_async();                    // BsT[buf^1] landed
        __syncthreads();                 // publish As + BsT[buf^1]
        buf ^= 1;
    }

    // epilogue
#pragma unroll
    for (int i = 0; i < 2; ++i) {
#pragma unroll
        for (int n = 0; n < 4; ++n) {
            const int col = n0 + wn + n * 16 + ml;
            const float bvv = bias[col];
            const int row0 = m0 + wm + i * 16 + rb;
            if (MODE == 3) {
                // f16 transposed per head: [b][h][dh][nk], 8 consecutive nk -> b128
                int bb = row0 >> 10, nk0 = row0 & 1023;
                int hh = col >> 6,  dd  = col & 63;
                v8h pk;
#pragma unroll
                for (int r = 0; r < 8; ++r) pk[r] = (_Float16)(acc[i][n][r] + bvv);
                *(v8h*)(C16 + (((size_t)bb * NH + hh) * DH + dd) * NK + nk0) = pk;
            } else {
#pragma unroll
                for (int r = 0; r < 8; ++r) {
                    int row = row0 + r;
                    float v = acc[i][n][r] + bvv;
                    if (MODE == 1) {
                        float x = X[(size_t)row * DIM + col];
                        v = (v > 0.f ? v : 0.f) + x;
                    }
                    if (MODE == 2) C16[(size_t)row * DIM + col] = (_Float16)v;
                    else           C  [(size_t)row * DIM + col] = v;
                }
            }
        }
    }
}

// ---------------------------------------------------------------------------
// Flash attention per (b,h): out = q + softmax(mask(q k^T / 8)) v
// K (f16 row-major) and V (f16 [b][h][dh][nk]) staged into LDS with
// GLOBAL_LOAD_ASYNC_TO_LDS_B128 (pure DMA, no VALU staging work).
// ---------------------------------------------------------------------------
__global__ __launch_bounds__(128)
void attn_kernel(const float* __restrict__ q32, const _Float16* __restrict__ k16,
                 const _Float16* __restrict__ v16T, const int* __restrict__ mask,
                 float* __restrict__ out) {
    constexpr int CK = 64;
    __shared__ __attribute__((aligned(16))) _Float16 Ks [CK][DH + 8]; // 144B stride
    __shared__ __attribute__((aligned(16))) _Float16 VsT[DH][CK + 8]; // 144B stride
    __shared__ __attribute__((aligned(16))) _Float16 Ps[4][16][CK + 8];

    const int tid  = threadIdx.x;
    const int wave = tid >> 5;
    const int lane = tid & 31;
    const int bh = blockIdx.y;
    const int b  = bh / NH, h = bh % NH;
    const int q0 = blockIdx.x * 64 + wave * 16;

    const size_t qbase = ((size_t)b * NQ + q0) * DIM + h * DH;
    const _Float16* kg = k16 + (size_t)b * NK * DIM + h * DH;   // [nk][512]+head ofs
    const _Float16* vg = v16T + (size_t)bh * DH * NK;           // [dh][nk]

    const int ml  = lane & 15;
    const int akb = (lane < 16) ? 0 : 8;
    const int bkb = (lane < 16) ? 0 : 8;
    const int rb  = (lane < 16) ? 0 : 8;

    // Q A-fragments with 1/sqrt(DH) folded in, f32 -> f16
    const float scale = 0.125f;
    v16h aq[2];
#pragma unroll
    for (int t = 0; t < 2; ++t) {
        const float* qr = q32 + qbase + (size_t)ml * DIM + t * 32;
#pragma unroll
        for (int j = 0; j < 4; ++j) {
            aq[t][2 * j]         = (_Float16)(qr[akb + 2 * j] * scale);
            aq[t][2 * j + 1]     = (_Float16)(qr[akb + 2 * j + 1] * scale);
            aq[t][8 + 2 * j]     = (_Float16)(qr[akb + 16 + 2 * j] * scale);
            aq[t][8 + 2 * j + 1] = (_Float16)(qr[akb + 16 + 2 * j + 1] * scale);
        }
    }

    v8f oacc[4] = {};
    float mrow[8], lrow[8];
#pragma unroll
    for (int r = 0; r < 8; ++r) { mrow[r] = -3.0e38f; lrow[r] = 0.f; }

    for (int kv0 = 0; kv0 < NK; kv0 += CK) {
        // stage K [kv][dh] and V [dh][kv] via async DMA (8KB each)
#pragma unroll
        for (int i = 0; i < 4; ++i) {
            int idx = tid + i * 128;
            int r = idx >> 3, s = (idx & 7) * 8;
            async_copy_b128(kg + (size_t)(kv0 + r) * DIM + s, &Ks[r][s]);
        }
#pragma unroll
        for (int i = 0; i < 4; ++i) {
            int idx = tid + i * 128;
            int d = idx >> 3, s = (idx & 7) * 8;
            async_copy_b128(vg + (size_t)d * NK + kv0 + s, &VsT[d][s]);
        }
        wait_async();
        __syncthreads();

        // S = (Q/8) @ K^T : 4 kv tiles x 2 K-steps, kk-outer
        v8f sacc[4] = {};
#pragma unroll
        for (int kk = 0; kk < 2; ++kk) {
            v16h bk[4];
#pragma unroll
            for (int t = 0; t < 4; ++t) {
                const _Float16* kr = &Ks[t * 16 + ml][kk * 32];
#pragma unroll
                for (int j = 0; j < 8; ++j) {
                    bk[t][2 * j]     = kr[bkb + 2 * j + (j >= 4 ? 8 : 0)];
                    bk[t][2 * j + 1] = kr[bkb + 2 * j + 1 + (j >= 4 ? 8 : 0)];
                }
            }
#pragma unroll
            for (int t = 0; t < 4; ++t)
                sacc[t] = __builtin_amdgcn_wmma_f32_16x16x32_f16(
                    false, aq[kk], false, bk[t], (short)0, sacc[t], false, false);
        }

        // online softmax over this 64-wide chunk
        int mk[4];
#pragma unroll
        for (int t = 0; t < 4; ++t) mk[t] = mask[b * NK + kv0 + t * 16 + ml];
#pragma unroll
        for (int r = 0; r < 8; ++r) {
            float s[4];
#pragma unroll
            for (int t = 0; t < 4; ++t)
                s[t] = mk[t] ? sacc[t][r] : NEGV;
            float mx = fmaxf(fmaxf(s[0], s[1]), fmaxf(s[2], s[3]));
            mx = half16_max(mx);
            float mnew = fmaxf(mrow[r], mx);
            float a = __expf(mrow[r] - mnew);
            float p[4], rs = 0.f;
#pragma unroll
            for (int t = 0; t < 4; ++t) { p[t] = __expf(s[t] - mnew); rs += p[t]; }
            rs = half16_sum(rs);
            lrow[r] = lrow[r] * a + rs;
            mrow[r] = mnew;
#pragma unroll
            for (int n = 0; n < 4; ++n) oacc[n][r] *= a;
#pragma unroll
            for (int t = 0; t < 4; ++t)
                Ps[wave][rb + r][t * 16 + ml] = (_Float16)p[t];
        }
        __syncthreads();

        // P A-fragments (K = 64 -> 2 frags)
        v16h ap[2];
#pragma unroll
        for (int kk = 0; kk < 2; ++kk) {
            const _Float16* pr = &Ps[wave][ml][kk * 32];
#pragma unroll
            for (int j = 0; j < 4; ++j) {
                ap[kk][2 * j]         = pr[akb + 2 * j];
                ap[kk][2 * j + 1]     = pr[akb + 2 * j + 1];
                ap[kk][8 + 2 * j]     = pr[akb + 16 + 2 * j];
                ap[kk][8 + 2 * j + 1] = pr[akb + 16 + 2 * j + 1];
            }
        }

        // O += P @ V : 4 dh tiles x 2 K-steps, kk-outer
#pragma unroll
        for (int kk = 0; kk < 2; ++kk) {
            v16h bv[4];
#pragma unroll
            for (int n = 0; n < 4; ++n) {
                const _Float16* vr = &VsT[n * 16 + ml][kk * 32];
#pragma unroll
                for (int j = 0; j < 8; ++j) {
                    bv[n][2 * j]     = vr[bkb + 2 * j + (j >= 4 ? 8 : 0)];
                    bv[n][2 * j + 1] = vr[bkb + 2 * j + 1 + (j >= 4 ? 8 : 0)];
                }
            }
#pragma unroll
            for (int n = 0; n < 4; ++n)
                oacc[n] = __builtin_amdgcn_wmma_f32_16x16x32_f16(
                    false, ap[kk], false, bv[n], (short)0, oacc[n], false, false);
        }
        __syncthreads();
    }

    // epilogue: out = q + O / l
    float rl[8];
#pragma unroll
    for (int r = 0; r < 8; ++r) rl[r] = 1.0f / lrow[r];
#pragma unroll
    for (int n = 0; n < 4; ++n) {
#pragma unroll
        for (int r = 0; r < 8; ++r) {
            size_t idx = qbase + (size_t)(rb + r) * DIM + n * 16 + ml;
            out[idx] = q32[idx] + oacc[n][r] * rl[r];
        }
    }
}

// ---------------------------------------------------------------------------
// LayerNorm over 512 cols; one wave per row, 8 rows per block.
// ---------------------------------------------------------------------------
__global__ __launch_bounds__(256)
void layernorm_kernel(const float* __restrict__ in, const float* __restrict__ g,
                      const float* __restrict__ bta, float* __restrict__ out) {
    const int wave = threadIdx.x >> 5, lane = threadIdx.x & 31;
    const int row = blockIdx.x * 8 + wave;
    const float* x = in + (size_t)row * DIM;
    float v[16];
    float s = 0.f;
#pragma unroll
    for (int i = 0; i < 16; ++i) { v[i] = x[lane + i * 32]; s += v[i]; }
#pragma unroll
    for (int off = 1; off < 32; off <<= 1) s += __shfl_xor(s, off, 32);
    float mu = s * (1.f / DIM);
    float q = 0.f;
#pragma unroll
    for (int i = 0; i < 16; ++i) { float d = v[i] - mu; q += d * d; }
#pragma unroll
    for (int off = 1; off < 32; off <<= 1) q += __shfl_xor(q, off, 32);
    float rstd = rsqrtf(q * (1.f / DIM) + 1e-5f);
    float* y = out + (size_t)row * DIM;
#pragma unroll
    for (int i = 0; i < 16; ++i) {
        int c = lane + i * 32;
        y[c] = (v[i] - mu) * rstd * g[c] + bta[c];
    }
}

// ---------------------------------------------------------------------------
// launch
// ---------------------------------------------------------------------------
extern "C" void kernel_launch(void* const* d_in, const int* in_sizes, int n_in,
                              void* d_out, int out_size, void* d_ws, size_t ws_size,
                              hipStream_t stream) {
    const float* Q    = (const float*)d_in[0];
    const float* K    = (const float*)d_in[1];
    const int*   mask = (const int*)  d_in[2];
    const float* Wq   = (const float*)d_in[3];
    const float* bq   = (const float*)d_in[4];
    const float* Wk   = (const float*)d_in[5];
    const float* bk   = (const float*)d_in[6];
    const float* Wv   = (const float*)d_in[7];
    const float* bv   = (const float*)d_in[8];
    const float* Wo   = (const float*)d_in[9];
    const float* bo   = (const float*)d_in[10];
    const float* g0   = (const float*)d_in[11];
    const float* b0   = (const float*)d_in[12];
    const float* g1   = (const float*)d_in[13];
    const float* b1   = (const float*)d_in[14];

    const size_t MB  = (size_t)B_SZ * NQ * DIM * sizeof(float);     // 16 MiB
    const size_t HB  = (size_t)B_SZ * NQ * DIM * sizeof(_Float16);  // 8 MiB
    char* ws = (char*)d_ws;
    float*    q32  = (float*)   (ws + 0 * MB);
    float*    attn = (float*)   (ws + 1 * MB);
    float*    X    = (float*)   (ws + 2 * MB);
    float*    Y    = (float*)   (ws + 3 * MB);
    _Float16* k16  = (_Float16*)(ws + 4 * MB);
    _Float16* v16T = (_Float16*)(ws + 4 * MB + HB);
    _Float16* W16  = (_Float16*)(ws + 4 * MB + 2 * HB);
    _Float16* WqT = W16 + 0 * (size_t)DIM * DIM;
    _Float16* WkT = W16 + 1 * (size_t)DIM * DIM;
    _Float16* WvT = W16 + 2 * (size_t)DIM * DIM;
    _Float16* WoT = W16 + 3 * (size_t)DIM * DIM;

    const int nW = DIM * DIM;
    cvt_transpose_f16<<<(nW + 255) / 256, 256, 0, stream>>>(Wq, WqT);
    cvt_transpose_f16<<<(nW + 255) / 256, 256, 0, stream>>>(Wk, WkT);
    cvt_transpose_f16<<<(nW + 255) / 256, 256, 0, stream>>>(Wv, WvT);
    cvt_transpose_f16<<<(nW + 255) / 256, 256, 0, stream>>>(Wo, WoT);

    const int Mrows = B_SZ * NQ; // 8192
    dim3 gg(Mrows / 128, DIM / 128);
    gemm512<0><<<gg, 256, 0, stream>>>(Q, WqT, bq, nullptr, q32, nullptr);
    gemm512<2><<<gg, 256, 0, stream>>>(K, WkT, bk, nullptr, nullptr, k16);
    gemm512<3><<<gg, 256, 0, stream>>>(K, WvT, bv, nullptr, nullptr, v16T);

    attn_kernel<<<dim3(NQ / 64, B_SZ * NH), 128, 0, stream>>>(q32, k16, v16T, mask, attn);

    layernorm_kernel<<<Mrows / 8, 256, 0, stream>>>(attn, g0, b0, X);
    gemm512<1><<<gg, 256, 0, stream>>>(X, WoT, bo, X, Y, nullptr);
    layernorm_kernel<<<Mrows / 8, 256, 0, stream>>>(Y, g1, b1, (float*)d_out);
}
